// PhylogeneticGNN_21844203668127
// MI455X (gfx1250) — matrix operand order repaired and checked by reading, not compiled, not run on hardware
//
#include <hip/hip_runtime.h>
#include <hip/hip_bf16.h>

typedef _Float16 v16h __attribute__((ext_vector_type(16)));
typedef float    v8f  __attribute__((ext_vector_type(8)));

// ---------------------------------------------------------------------------
// Degree / norm precompute
// ---------------------------------------------------------------------------
__global__ void deg_init_kernel(float* __restrict__ deg, int N) {
    int i = blockIdx.x * blockDim.x + threadIdx.x;
    if (i < N) deg[i] = 1.0f;  // self-loop weight
}

__global__ void deg_scatter_kernel(const int* __restrict__ col,
                                   const float* __restrict__ ew,
                                   float* __restrict__ deg, int E) {
    int e = blockIdx.x * blockDim.x + threadIdx.x;
    if (e < E) {
        __hip_atomic_fetch_add(&deg[col[e]], ew[e],
                               __ATOMIC_RELAXED, __HIP_MEMORY_SCOPE_AGENT);
    }
}

__global__ void dinv_kernel(float* __restrict__ deg, int N) {
    int i = blockIdx.x * blockDim.x + threadIdx.x;
    if (i < N) {
        float d = deg[i];
        deg[i] = (d > 0.0f) ? rsqrtf(d) : 0.0f;  // in place: deg -> dinv
    }
}

__global__ void norm_kernel(const int* __restrict__ row, const int* __restrict__ col,
                            const float* __restrict__ ew, const float* __restrict__ dinv,
                            float* __restrict__ norm, int E, int NT /* = E+N */) {
    int e = blockIdx.x * blockDim.x + threadIdx.x;
    if (e >= NT) return;
    if (e < E) {
        norm[e] = dinv[row[e]] * ew[e] * dinv[col[e]];
    } else {
        float d = dinv[e - E];
        norm[e] = d * d;  // self-loop, weight 1
    }
}

// ---------------------------------------------------------------------------
// Weight pack: W[K,Nc] (f32, row-major) -> per-(colTile, kstep, lane) 16-f16
// fragments in the exact 16-bit B 32x16 WMMA layout, zero-padded past K.
// Fragment index = (colTile*numK + kk)*32 + lane; 32 contiguous bytes each.
// ---------------------------------------------------------------------------
__global__ void pack_w_kernel(const float* __restrict__ W, _Float16* __restrict__ Bp,
                              int K, int Nc, int numK) {
    int gid = blockIdx.x * blockDim.x + threadIdx.x;
    int colTiles = Nc >> 4;
    int total = colTiles * numK * 32;
    if (gid >= total) return;
    int lane = gid & 31;
    int kk   = (gid >> 5) % numK;
    int lmod = lane & 15, lhi = lane >> 4;
    int ct   = gid / (numK << 5);
    int ncol = ct * 16 + lmod;
    _Float16* dst = Bp + (size_t)gid * 16;
#pragma unroll
    for (int i = 0; i < 16; ++i) {
        int k = (kk << 5) + ((i >> 3) << 4) + (lhi << 3) + (i & 7);
        dst[i] = (k < K) ? (_Float16)W[(size_t)k * Nc + ncol] : (_Float16)0.0f;
    }
}

// ---------------------------------------------------------------------------
// WMMA GEMM: C[M,Nc] = opt_relu( relu_in?(A[M,K]) @ W[K,Nc] + bias )
// One wave computes a 16x32 C tile: one A fragment feeds TWO v_wmma ops with
// two packed-B fragments (halves A traffic / cvt work per WMMA). Main K loop
// is guard-free with b128 loads; only a final partial 32-step (K=100 case)
// takes the guarded tail. Requires colTiles even (always true here).
// ---------------------------------------------------------------------------
template <bool IN_RELU>
__global__ __launch_bounds__(128)
void gemm_wmma_kernel(const float* __restrict__ A, const _Float16* __restrict__ Bp,
                      const float* __restrict__ bias, float* __restrict__ C,
                      int M, int K, int Nc, int numK, int outRelu) {
    const int lane   = threadIdx.x & 31;
    const int waveId = blockIdx.x * 4 + (threadIdx.x >> 5);
    const int colPairs = Nc >> 5;                 // 32-wide column pairs
    const int rowTile  = waveId / colPairs;
    const int colTile0 = (waveId % colPairs) * 2; // first of two 16-col tiles
    if (rowTile * 16 >= M) return;                // wave-uniform

    const int lmod = lane & 15;
    const int lhi  = lane >> 4;

    int m = rowTile * 16 + lmod;
    if (m >= M) m = M - 1;                        // clamp (results discarded)
    const float* __restrict__ Arow = A + (size_t)m * K;
    const v16h* __restrict__ Bfrag =
        (const v16h*)Bp + (size_t)colTile0 * numK * 32 + lane;
    const size_t bStride = (size_t)numK * 32;     // v16h elements per col tile
    const int ncol0 = colTile0 * 16 + lmod;

    v8f acc0 = {};
    v8f acc1 = {};
    const int kFull = K >> 5;                     // full 32-wide steps

#pragma unroll 2
    for (int kk = 0; kk < kFull; ++kk) {
        const int k0 = kk << 5;
        __builtin_prefetch(Arow + k0 + 64, 0, 3);

        // A fragment: two contiguous 8-float runs per lane -> 4x b128.
        const float4* __restrict__ Aq = (const float4*)(Arow + k0 + (lhi << 3));
        float4 a0 = Aq[0];
        float4 a1 = Aq[1];
        float4 a2 = Aq[4];   // +16 floats
        float4 a3 = Aq[5];
        if (IN_RELU) {
            a0.x = fmaxf(a0.x, 0.f); a0.y = fmaxf(a0.y, 0.f); a0.z = fmaxf(a0.z, 0.f); a0.w = fmaxf(a0.w, 0.f);
            a1.x = fmaxf(a1.x, 0.f); a1.y = fmaxf(a1.y, 0.f); a1.z = fmaxf(a1.z, 0.f); a1.w = fmaxf(a1.w, 0.f);
            a2.x = fmaxf(a2.x, 0.f); a2.y = fmaxf(a2.y, 0.f); a2.z = fmaxf(a2.z, 0.f); a2.w = fmaxf(a2.w, 0.f);
            a3.x = fmaxf(a3.x, 0.f); a3.y = fmaxf(a3.y, 0.f); a3.z = fmaxf(a3.z, 0.f); a3.w = fmaxf(a3.w, 0.f);
        }
        v16h a;
        a[0]  = (_Float16)a0.x; a[1]  = (_Float16)a0.y; a[2]  = (_Float16)a0.z; a[3]  = (_Float16)a0.w;
        a[4]  = (_Float16)a1.x; a[5]  = (_Float16)a1.y; a[6]  = (_Float16)a1.z; a[7]  = (_Float16)a1.w;
        a[8]  = (_Float16)a2.x; a[9]  = (_Float16)a2.y; a[10] = (_Float16)a2.z; a[11] = (_Float16)a2.w;
        a[12] = (_Float16)a3.x; a[13] = (_Float16)a3.y; a[14] = (_Float16)a3.z; a[15] = (_Float16)a3.w;

        // Two B fragments (consecutive col tiles): contiguous 32B loads.
        v16h b0 = Bfrag[(size_t)kk * 32];
        v16h b1 = Bfrag[(size_t)kk * 32 + bStride];

        acc0 = __builtin_amdgcn_wmma_f32_16x16x32_f16(
            false, a, false, b0, (short)0, acc0, false, false);
        acc1 = __builtin_amdgcn_wmma_f32_16x16x32_f16(
            false, a, false, b1, (short)0, acc1, false, false);
    }

    // Tail (only for K % 32 != 0, e.g. K = 100): guarded scalar A loads.
    if ((kFull << 5) < K) {
        const int k0 = kFull << 5;
        v16h a;
#pragma unroll
        for (int i = 0; i < 16; ++i) {
            int k = k0 + ((i >> 3) << 4) + (lhi << 3) + (i & 7);
            float v = (k < K) ? Arow[k] : 0.0f;
            if (IN_RELU) v = fmaxf(v, 0.0f);
            a[i] = (_Float16)v;
        }
        v16h b0 = Bfrag[(size_t)kFull * 32];             // zero-padded at pack
        v16h b1 = Bfrag[(size_t)kFull * 32 + bStride];
        acc0 = __builtin_amdgcn_wmma_f32_16x16x32_f16(
            false, a, false, b0, (short)0, acc0, false, false);
        acc1 = __builtin_amdgcn_wmma_f32_16x16x32_f16(
            false, a, false, b1, (short)0, acc1, false, false);
    }

    const float bv0 = bias ? bias[ncol0]      : 0.0f;
    const float bv1 = bias ? bias[ncol0 + 16] : 0.0f;
#pragma unroll
    for (int r = 0; r < 8; ++r) {
        int mm = rowTile * 16 + (lhi << 3) + r;  // VGPR r: M=r (lanes 0-15), 8+r (16-31)
        if (mm < M) {
            float* __restrict__ Crow = C + (size_t)mm * Nc;
            float v0 = acc0[r] + bv0;
            float v1 = acc1[r] + bv1;
            if (outRelu) { v0 = fmaxf(v0, 0.0f); v1 = fmaxf(v1, 0.0f); }
            Crow[ncol0]      = v0;
            Crow[ncol0 + 16] = v1;
        }
    }
}

// ---------------------------------------------------------------------------
// out[n][j] = bias[j]  (initialize conv output with bias before scatter)
// ---------------------------------------------------------------------------
__global__ void fill_bias_kernel(float* __restrict__ out, const float* __restrict__ bias,
                                 long total, int H) {
    long idx = (long)blockIdx.x * blockDim.x + threadIdx.x;
    if (idx < total) out[idx] = bias[idx % H];
}

// ---------------------------------------------------------------------------
// Scatter: out[col[e]] += norm[e] * tmp[row[e]]   (self-loops at e >= E)
// ---------------------------------------------------------------------------
__global__ __launch_bounds__(256)
void scatter_msg_kernel(const float* __restrict__ tmp, float* __restrict__ out,
                        const int* __restrict__ row, const int* __restrict__ col,
                        const float* __restrict__ norm, int E, int NT, int H) {
    const int H4  = H >> 2;
    const int epb = 256 / H4;
    int e = blockIdx.x * epb + threadIdx.x / H4;
    int j = threadIdx.x % H4;
    if (e >= NT) return;
    int r, c;
    if (e < E) { r = row[e]; c = col[e]; }
    else       { r = e - E;  c = r; }
    float w = norm[e];
    float4 v = *(const float4*)(tmp + (size_t)r * H + (size_t)j * 4);
    float* o = out + (size_t)c * H + (size_t)j * 4;
    __hip_atomic_fetch_add(o + 0, w * v.x, __ATOMIC_RELAXED, __HIP_MEMORY_SCOPE_AGENT);
    __hip_atomic_fetch_add(o + 1, w * v.y, __ATOMIC_RELAXED, __HIP_MEMORY_SCOPE_AGENT);
    __hip_atomic_fetch_add(o + 2, w * v.z, __ATOMIC_RELAXED, __HIP_MEMORY_SCOPE_AGENT);
    __hip_atomic_fetch_add(o + 3, w * v.w, __ATOMIC_RELAXED, __HIP_MEMORY_SCOPE_AGENT);
}

// ---------------------------------------------------------------------------
// Head: only row t of the output projection matters.
// ---------------------------------------------------------------------------
__global__ __launch_bounds__(512)
void head_kernel(const float* __restrict__ hfin, const int* __restrict__ tsp,
                 const float* __restrict__ Wo1, const float* __restrict__ bo1,
                 const float* __restrict__ Wo2, const float* __restrict__ bo2,
                 const float* __restrict__ Wh1, const float* __restrict__ bh1,
                 const float* __restrict__ Wh2, const float* __restrict__ bh2,
                 const float* __restrict__ Wh3, const float* __restrict__ bh3,
                 float* __restrict__ out,
                 int H, int H2, int OUT, int D1, int D2, int D3) {
    __shared__ float s0[512];
    __shared__ float s1[512];
    const int tid = threadIdx.x;
    const int t = tsp[0];

    if (tid < H) s0[tid] = hfin[(size_t)t * H + tid];
    __syncthreads();

    if (tid < H2) {                       // relu(h @ Wo1 + bo1)
        float acc = bo1[tid];
        for (int k = 0; k < H; ++k) acc += s0[k] * Wo1[(size_t)k * H2 + tid];
        s1[tid] = fmaxf(acc, 0.0f);
    }
    __syncthreads();
    if (tid < OUT) {                      // @ Wo2 + bo2
        float acc = bo2[tid];
        for (int k = 0; k < H2; ++k) acc += s1[k] * Wo2[(size_t)k * OUT + tid];
        s0[tid] = acc;
    }
    __syncthreads();
    if (tid < D1) {                       // relu(t @ Wh1 + bh1)
        float acc = bh1[tid];
        for (int k = 0; k < OUT; ++k) acc += s0[k] * Wh1[(size_t)k * D1 + tid];
        s1[tid] = fmaxf(acc, 0.0f);
    }
    __syncthreads();
    if (tid < D2) {                       // relu(t @ Wh2 + bh2)
        float acc = bh2[tid];
        for (int k = 0; k < D1; ++k) acc += s1[k] * Wh2[(size_t)k * D2 + tid];
        s0[tid] = fmaxf(acc, 0.0f);
    }
    __syncthreads();
    if (tid < D3) {                       // t @ Wh3 + bh3  -> 5 outputs
        float acc = bh3[tid];
        for (int k = 0; k < D2; ++k) acc += s0[k] * Wh3[(size_t)k * D3 + tid];
        out[tid] = acc;
    }
}

// ---------------------------------------------------------------------------
// Host side
// ---------------------------------------------------------------------------
static inline void launch_gemm(const float* A, const float* W, const float* bias,
                               float* C, _Float16* Bp, int M, int K, int Nc,
                               int inRelu, int outRelu, hipStream_t s) {
    const int numK = (K + 31) / 32;
    const int colTiles = Nc / 16;
    const int packThreads = colTiles * numK * 32;
    pack_w_kernel<<<(packThreads + 255) / 256, 256, 0, s>>>(W, Bp, K, Nc, numK);

    const int rowTiles = (M + 15) / 16;
    const long waves = (long)rowTiles * (colTiles / 2);   // 16x32 per wave
    const int blocks = (int)((waves + 3) / 4);
    if (inRelu)
        gemm_wmma_kernel<true ><<<blocks, 128, 0, s>>>(A, Bp, bias, C, M, K, Nc, numK, outRelu);
    else
        gemm_wmma_kernel<false><<<blocks, 128, 0, s>>>(A, Bp, bias, C, M, K, Nc, numK, outRelu);
}

extern "C" void kernel_launch(void* const* d_in, const int* in_sizes, int n_in,
                              void* d_out, int out_size, void* d_ws, size_t ws_size,
                              hipStream_t stream) {
    const float* x   = (const float*)d_in[0];
    const int*   ei  = (const int*)d_in[1];   // [2, E]
    const float* ew  = (const float*)d_in[2];
    const int*   tsp = (const int*)d_in[3];
    const float* Wi  = (const float*)d_in[4];
    const float* bi  = (const float*)d_in[5];
    const float* Wg  = (const float*)d_in[6];  // [L, H, H]
    const float* bg  = (const float*)d_in[7];  // [L, H]
    const float* Wo1 = (const float*)d_in[8];
    const float* bo1 = (const float*)d_in[9];
    const float* Wo2 = (const float*)d_in[10];
    const float* bo2 = (const float*)d_in[11];
    const float* Wh1 = (const float*)d_in[12];
    const float* bh1 = (const float*)d_in[13];
    const float* Wh2 = (const float*)d_in[14];
    const float* bh2 = (const float*)d_in[15];
    const float* Wh3 = (const float*)d_in[16];
    const float* bh3 = (const float*)d_in[17];

    const int H    = in_sizes[5];            // 256
    const int F_IN = in_sizes[4] / H;        // 100
    const int N    = in_sizes[0] / F_IN;     // 50000
    const int E    = in_sizes[2];            // 300000
    const int H2   = in_sizes[9];            // 512
    const int OUT  = in_sizes[11];           // 512
    const int L    = in_sizes[7] / H;        // 3
    const int D1   = in_sizes[13];           // 256
    const int D2   = in_sizes[15];           // 128
    const int D3   = out_size;               // 5
    const int NT   = E + N;

    const int* row = ei;
    const int* col = ei + E;

    // Workspace layout
    float* hA   = (float*)d_ws;              // [N, H]
    float* hB   = hA + (size_t)N * H;        // [N, H]
    float* dinv = hB + (size_t)N * H;        // [N] (deg -> dinv in place)
    float* nrm  = dinv + N;                  // [E + N]
    _Float16* Bp = (_Float16*)(nrm + NT);    // packed weights (<= 128 KB)

    // 1) degree / norm
    deg_init_kernel<<<(N + 255) / 256, 256, 0, stream>>>(dinv, N);
    deg_scatter_kernel<<<(E + 255) / 256, 256, 0, stream>>>(col, ew, dinv, E);
    dinv_kernel<<<(N + 255) / 256, 256, 0, stream>>>(dinv, N);
    norm_kernel<<<(NT + 255) / 256, 256, 0, stream>>>(row, col, ew, dinv, nrm, E, NT);

    // 2) h = relu(x @ Wi + bi)
    launch_gemm(x, Wi, bi, hA, Bp, N, F_IN, H, /*inRelu=*/0, /*outRelu=*/1, stream);

    // 3) GCN layers: tmp = relu_in(h) @ Wg[l]; out = bias; scatter(norm * tmp)
    const long totalNH = (long)N * H;
    const int  epb = 256 / (H >> 2);
    for (int l = 0; l < L; ++l) {
        launch_gemm(hA, Wg + (size_t)l * H * H, nullptr, hB, Bp,
                    N, H, H, /*inRelu=*/(l > 0), /*outRelu=*/0, stream);
        fill_bias_kernel<<<(int)((totalNH + 255) / 256), 256, 0, stream>>>(
            hA, bg + (size_t)l * H, totalNH, H);
        scatter_msg_kernel<<<(NT + epb - 1) / epb, 256, 0, stream>>>(
            hB, hA, row, col, nrm, E, NT, H);
    }

    // 4) output projection + head, only for row target_species_idx
    head_kernel<<<1, 512, 0, stream>>>(hA, tsp,
                                       Wo1, bo1, Wo2, bo2,
                                       Wh1, bh1, Wh2, bh2, Wh3, bh3,
                                       (float*)d_out, H, H2, OUT, D1, D2, D3);
}
